// MultiHeadAttention_19971597926572
// MI455X (gfx1250) — matrix operand order repaired and checked by reading, not compiled, not run on hardware
//
#include <hip/hip_runtime.h>
#include <stdint.h>

// Problem constants (match reference)
#define B_  4096
#define T_  128
#define C_  128
#define H_  4
#define D_  32
#define LDW 136           // LDS row stride in bf16 elements (272B, 16B-aligned rows)
#define SZ  (128 * LDW)   // one 128-row staging buffer

typedef __attribute__((ext_vector_type(16))) __bf16 v16bf;
typedef __attribute__((ext_vector_type(8)))  float  v8f;

__device__ __forceinline__ __bf16 f2bf(float f) { return (__bf16)f; }  // v_cvt_pk_bf16_f32

// ---- CDNA5 WMMA bf16 fragment loaders (ISA 7.12.2 layouts) ----

// A fragment (16x32 bf16), row-major source, stride LDW.
// lane: m = lane&15; halves = K[kb..kb+7] ++ K[kb+16..kb+23], kb = (lane>=16)?8:0
__device__ __forceinline__ v16bf load_a(const __bf16* base, int row0, int col0, int lane) {
  const int m  = lane & 15;
  const int kb = (lane >> 4) << 3;
  const __bf16* r = base + (row0 + m) * LDW + col0 + kb;
  v16bf t;
#pragma unroll
  for (int j = 0; j < 8; ++j) { t[j] = r[j]; t[8 + j] = r[16 + j]; }
  return t;
}

// B fragment (32x16 bf16): lane holds column n = lane&15; halves = K[kb..kb+7] ++ K[kb+16..kb+23], kb = (lane>=16)?8:0
// Source holds B^T row-major (K along columns).
__device__ __forceinline__ v16bf load_b_row(const __bf16* base, int row0, int col0, int lane) {
  const int n  = lane & 15;
  const int kb = (lane >> 4) << 3;
  const __bf16* p = base + (row0 + n) * LDW + col0 + kb;
  v16bf t;
#pragma unroll
  for (int j = 0; j < 8; ++j) { t[j] = p[j]; t[8 + j] = p[16 + j]; }
  return t;
}

// D fragment (16x16 fp32) -> bf16 LDS, row-major: row = v + 8*(lane>=16), col = lane&15
__device__ __forceinline__ void store_d(__bf16* base, int row0, int col0, int lane, const v8f& c) {
  const int n  = lane & 15;
  const int rb = (lane >> 4) << 3;
  __bf16* p = base + (row0 + rb) * LDW + col0 + n;
#pragma unroll
  for (int v = 0; v < 8; ++v) p[v * LDW] = f2bf(c[v]);
}

// D fragment -> bf16 LDS TRANSPOSED (dst[col][row]); per-lane contiguous 16B store.
__device__ __forceinline__ void store_d_T(__bf16* base, int row0, int col0, int lane, const v8f& c) {
  const int n  = lane & 15;
  const int rb = (lane >> 4) << 3;
  __bf16* p = base + (col0 + n) * LDW + row0 + rb;
#pragma unroll
  for (int v = 0; v < 8; ++v) p[v] = f2bf(c[v]);
}

__device__ __forceinline__ v8f wmma_bf16(v16bf a, v16bf b, v8f c) {
  // (neg_a, A, neg_b, B, c_mod, C, reuse_a, reuse_b)
  return __builtin_amdgcn_wmma_f32_16x16x32_bf16(false, a, false, b, (short)0, c, false, false);
}

__global__ __launch_bounds__(256, 1)
void mha_fused_kernel(const float* __restrict__ x,
                      const float* __restrict__ Wq,
                      const float* __restrict__ Wk,
                      const float* __restrict__ Wv,
                      const float* __restrict__ Wp,
                      const float* __restrict__ bp,
                      float* __restrict__ out) {
  __shared__ __align__(16) __bf16 smem[7 * SZ];
  __bf16* sX   = smem + 0 * SZ;   // phase 1-2: X bf16
  __bf16* sO   = smem + 0 * SZ;   // phase 4-5: attention output (alias, X dead)
  __bf16* sWp  = smem + 1 * SZ;   // phase 2: Wq^T | phase 3-5: Wp (row-major)
  __bf16* sP   = smem + 2 * SZ;   // phase 2: Wk^T | phase 4: per-wave P scratch
  __bf16* sQ   = smem + 4 * SZ;
  __bf16* sK   = smem + 5 * SZ;
  __bf16* sVT  = smem + 6 * SZ;   // V stored transposed: sVT[d][s]

  const int b    = blockIdx.x;
  const int tid  = threadIdx.x;
  const int wave = tid >> 5;
  const int lane = tid & 31;

  // Warm Wp lines (used in phase 3) -- gfx1250 global_prefetch_b8
  __builtin_prefetch(Wp + tid * 64, 0, 1);

  // ---- phase 1: stage X[b] (row-major) and Wq/Wk/Wv (transposed) as bf16 ----
  {
    const float4* x4 = (const float4*)(x + (size_t)b * (T_ * C_));
    for (int i = tid; i < (T_ * C_) / 4; i += 256) {
      float4 f = x4[i];
      const int row = (i * 4) >> 7;
      const int col = (i * 4) & 127;
      __bf16* p = sX + row * LDW + col;
      p[0] = f2bf(f.x); p[1] = f2bf(f.y); p[2] = f2bf(f.z); p[3] = f2bf(f.w);
    }
    // Weights: W[h][c][d] -> sWT[h*32+d][c]  (combined column j = h*32+d)
    for (int i = tid; i < 3 * (H_ * C_ * D_) / 4; i += 256) {
      const int sel = i >> 12;            // 4096 float4 per weight matrix
      const int r   = i & 4095;
      const float4* w4 = (const float4*)((sel == 0) ? Wq : (sel == 1) ? Wk : Wv);
      float4 f = w4[r];
      const int e = r * 4;                // element index within [H][C][D]
      const int h = e >> 12;
      const int c = (e >> 5) & 127;
      const int j = h * 32 + (e & 31);
      __bf16* dst = smem + (1 + sel) * SZ;
      dst[(j + 0) * LDW + c] = f2bf(f.x);
      dst[(j + 1) * LDW + c] = f2bf(f.y);
      dst[(j + 2) * LDW + c] = f2bf(f.z);
      dst[(j + 3) * LDW + c] = f2bf(f.w);
    }
  }
  __syncthreads();

  // ---- phase 2: Q/K/V = X · W (three compile-time-sel loops: no exec divergence) ----
  {
    const __bf16* WqT = smem + 1 * SZ;
    for (int t = wave; t < 64; t += 8) {
      const int rt = t >> 3, ct = t & 7;
      v8f acc = {};
#pragma unroll
      for (int kk = 0; kk < 4; ++kk)
        acc = wmma_bf16(load_a(sX, rt * 16, kk * 32, lane),
                        load_b_row(WqT, ct * 16, kk * 32, lane), acc);
      store_d(sQ, rt * 16, ct * 16, lane, acc);
    }
    const __bf16* WkT = smem + 2 * SZ;
    for (int t = wave; t < 64; t += 8) {
      const int rt = t >> 3, ct = t & 7;
      v8f acc = {};
#pragma unroll
      for (int kk = 0; kk < 4; ++kk)
        acc = wmma_bf16(load_a(sX, rt * 16, kk * 32, lane),
                        load_b_row(WkT, ct * 16, kk * 32, lane), acc);
      store_d(sK, rt * 16, ct * 16, lane, acc);
    }
    const __bf16* WvT = smem + 3 * SZ;
    for (int t = wave; t < 64; t += 8) {
      const int rt = t >> 3, ct = t & 7;
      v8f acc = {};
#pragma unroll
      for (int kk = 0; kk < 4; ++kk)
        acc = wmma_bf16(load_a(sX, rt * 16, kk * 32, lane),
                        load_b_row(WvT, ct * 16, kk * 32, lane), acc);
      store_d_T(sVT, rt * 16, ct * 16, lane, acc);   // V transposed, contiguous 16B store
    }
  }
  __syncthreads();

  // ---- phase 3: stage Wp fp32 -> bf16 (row-major) into slot 1 (Wq^T dead) ----
  {
    const float4* w4 = (const float4*)Wp;
    for (int i = tid; i < (C_ * C_) / 4; i += 256) {
      float4 f = w4[i];
      const int row = (i * 4) >> 7;
      const int col = (i * 4) & 127;
      __bf16* p = sWp + row * LDW + col;
      p[0] = f2bf(f.x); p[1] = f2bf(f.y); p[2] = f2bf(f.z); p[3] = f2bf(f.w);
    }
  }
  // sWp is only read after the post-attention barrier; no barrier needed here.

  // ---- phase 4: attention per (head, 16-row tile): Q·K^T, softmax, P·V ----
  const float scale = 0.08838834764831845f;   // C^-0.5 (reference scales by EMBED_DIM)
  __bf16* sPw = sP + wave * (16 * LDW);
  for (int pr = wave; pr < H_ * 8; pr += 8) {
    const int h  = pr >> 3;
    const int rt = pr & 7;

    // S = Q · K^T  (K-dim = D = 32 -> single WMMA per 16x16 tile)
    v16bf qa = load_a(sQ, rt * 16, h * 32, lane);
    v8f s[8];
#pragma unroll
    for (int ct = 0; ct < 8; ++ct) {
      v16bf kb = load_b_row(sK, ct * 16, h * 32, lane);    // B[d][s] = K[s][d]
      v8f z = {};
      s[ct] = wmma_bf16(qa, kb, z);
    }

    // scale + causal mask + row softmax (C-frag: row = v + 8*(lane>=16), col = ct*16 + n)
    const int n  = lane & 15;
    const int rb = (lane >> 4) << 3;
    float rmax[8], rsum[8];
#pragma unroll
    for (int v = 0; v < 8; ++v) rmax[v] = -3.0e38f;
#pragma unroll
    for (int ct = 0; ct < 8; ++ct) {
#pragma unroll
      for (int v = 0; v < 8; ++v) {
        const int M = rt * 16 + v + rb;
        const int N = ct * 16 + n;
        float val = s[ct][v] * scale;
        if (N > M) val = -3.0e38f;
        s[ct][v] = val;
        rmax[v] = fmaxf(rmax[v], val);
      }
    }
#pragma unroll
    for (int v = 0; v < 8; ++v) {
#pragma unroll
      for (int m = 1; m < 16; m <<= 1)
        rmax[v] = fmaxf(rmax[v], __shfl_xor(rmax[v], m, 32));
      rsum[v] = 0.0f;
    }
#pragma unroll
    for (int ct = 0; ct < 8; ++ct) {
#pragma unroll
      for (int v = 0; v < 8; ++v) {
        float e = __expf(s[ct][v] - rmax[v]);
        s[ct][v] = e;
        rsum[v] += e;
      }
    }
#pragma unroll
    for (int v = 0; v < 8; ++v) {
#pragma unroll
      for (int m = 1; m < 16; m <<= 1)
        rsum[v] += __shfl_xor(rsum[v], m, 32);
      rsum[v] = 1.0f / rsum[v];
    }
#pragma unroll
    for (int ct = 0; ct < 8; ++ct) {
#pragma unroll
      for (int v = 0; v < 8; ++v) s[ct][v] *= rsum[v];
      store_d(sPw, 0, ct * 16, lane, s[ct]);               // P -> per-wave bf16 scratch
    }

    // O = P · V  (K-dim = 128 -> 4 steps; N = D = 32 -> 2 column tiles)
#pragma unroll
    for (int ctd = 0; ctd < 2; ++ctd) {
      v8f acc = {};
#pragma unroll
      for (int kk = 0; kk < 4; ++kk) {
        v16bf a  = load_a(sPw, 0, kk * 32, lane);
        v16bf bv = load_b_row(sVT, h * 32 + ctd * 16, kk * 32, lane);  // V^T rows
        acc = wmma_bf16(a, bv, acc);
      }
      store_d(sO, rt * 16, h * 32 + ctd * 16, lane, acc);
    }
  }
  __syncthreads();

  // ---- phase 5: out = O · Wp^T + bp ----
  for (int tt = wave; tt < 64; tt += 8) {
    const int rt  = tt >> 3;
    const int ctc = tt & 7;
    v8f acc = {};
#pragma unroll
    for (int kk = 0; kk < 4; ++kk) {
      v16bf a  = load_a(sO, rt * 16, kk * 32, lane);
      v16bf bw = load_b_row(sWp, ctc * 16, kk * 32, lane); // B[j][c] = Wp[c][j]
      acc = wmma_bf16(a, bw, acc);
    }
    const int n  = lane & 15;
    const int rb = (lane >> 4) << 3;
    const float bias = bp[ctc * 16 + n];
    float* o = out + (size_t)b * (T_ * C_) + (size_t)(rt * 16 + rb) * C_ + ctc * 16 + n;
#pragma unroll
    for (int v = 0; v < 8; ++v) o[(size_t)v * C_] = acc[v] + bias;
  }
}

extern "C" void kernel_launch(void* const* d_in, const int* in_sizes, int n_in,
                              void* d_out, int out_size, void* d_ws, size_t ws_size,
                              hipStream_t stream) {
  (void)in_sizes; (void)n_in; (void)out_size; (void)d_ws; (void)ws_size;
  const float* x  = (const float*)d_in[0];
  const float* Wq = (const float*)d_in[1];
  const float* Wk = (const float*)d_in[2];
  const float* Wv = (const float*)d_in[3];
  const float* Wp = (const float*)d_in[4];
  const float* bp = (const float*)d_in[5];
  float* out = (float*)d_out;
  mha_fused_kernel<<<dim3(B_), dim3(256), 0, stream>>>(x, Wq, Wk, Wv, Wp, bp, out);
}
